// MultiHeadAttention_11854109737248
// MI455X (gfx1250) — compile-verified, matrix operand order checked by reading, not run on hardware
//
#include <hip/hip_runtime.h>
#include <hip/hip_bf16.h>

// ---------------- types / helpers ----------------
typedef __attribute__((ext_vector_type(16))) __bf16 v16bf;
typedef __attribute__((ext_vector_type(8)))  float  v8f;

union AFrag { v16bf v; unsigned int u[8]; };
union CFrag { v8f   v; float        f[8]; };

#define E_DIM   1024
#define T_DIM   2048
#define B_DIM   4
#define H_NUM   16
#define D_HEAD  64
#define R_ROWS  (T_DIM * B_DIM)      // 8192
#define SCALING 0.125f               // 64^-0.5

__device__ __forceinline__ unsigned short f2bf(float x) {
    union { float f; unsigned int u; } c; c.f = x;
    unsigned int u = c.u;
    u += 0x7FFFu + ((u >> 16) & 1u);       // round-to-nearest-even
    return (unsigned short)(u >> 16);
}

// Load a 16x32 bf16 WMMA A/B fragment from LDS where each logical row holds
// 32 contiguous bf16 K-values (viewed as uints). ISA wave32 layout:
// lanes 0-15: row=lane, K in {0..7,16..23}; lanes 16-31: row=lane-16, K+=8.
__device__ __forceinline__ void load_frag(AFrag& f, const unsigned int* lds,
                                          int strideU, int rowBase, int kBaseU,
                                          int lane) {
    const int rr = lane & 15;
    const int hi = (lane >> 4) << 2;       // +4 uints (= +8 bf16) for upper half-wave
    const unsigned int* p = lds + (size_t)(rowBase + rr) * strideU + kBaseU + hi;
#pragma unroll
    for (int j = 0; j < 8; ++j)
        f.u[j] = p[((j >> 2) << 3) + (j & 3)];   // pairs 0..3 then +8 uints (K=16..23)
}

__device__ __forceinline__ v8f wmma_bf16(const AFrag& a, const AFrag& b, const v8f& c) {
    return __builtin_amdgcn_wmma_f32_16x16x32_bf16(false, a.v, false, b.v,
                                                   (short)0, c, false, false);
}

// ---------------- kernel 1: f32 -> bf16 convert ----------------
__global__ __launch_bounds__(256) void cvt_bf16(const float* __restrict__ src,
                                                unsigned short* __restrict__ dst,
                                                int n) {
    int i = blockIdx.x * 256 + threadIdx.x;
    if (i < n) dst[i] = f2bf(src[i]);
}

// ---------------- kernel 2: tiled WMMA GEMM  out = X @ W^T + b ----------------
// MODE 0: Q -> bf16 [head][t][d], scaled          MODE 1: K -> bf16 [head][t][d]
// MODE 2: V -> bf16 transposed [head][d][t]       MODE 3: fp32 row-major to d_out
template <int MODE>
__global__ __launch_bounds__(256) void proj_gemm(const unsigned short* __restrict__ X,
                                                 const unsigned short* __restrict__ W,
                                                 const float* __restrict__ bias,
                                                 void* __restrict__ out,
                                                 float scale) {
    __shared__ unsigned short As[128 * 36];   // 128 rows x 32 bf16 (+4 pad)
    __shared__ unsigned short Bs[128 * 36];

    const int tid  = threadIdx.x;
    const int lane = tid & 31;
    const int w    = tid >> 5;
    const int wm   = w >> 2;                  // 0..1 : 64-row slice
    const int wn   = w & 3;                   // 0..3 : 32-col slice
    const int rBase = blockIdx.y * 128;
    const int oBase = blockIdx.x * 128;

    unsigned int* AsW = (unsigned int*)As;
    unsigned int* BsW = (unsigned int*)Bs;
    const unsigned int* AsU = (const unsigned int*)As;
    const unsigned int* BsU = (const unsigned int*)Bs;

    CFrag acc[4][2];
#pragma unroll
    for (int mt = 0; mt < 4; ++mt)
#pragma unroll
        for (int nt = 0; nt < 2; ++nt)
#pragma unroll
            for (int r = 0; r < 8; ++r) acc[mt][nt].f[r] = 0.0f;

    for (int kb = 0; kb < E_DIM / 32; ++kb) {
        const int kOff = kb * 32;
#pragma unroll
        for (int it = 0; it < 2; ++it) {
            const int c   = tid + it * 256;       // 512 16B chunks per tile
            const int row = c >> 2;
            const int cc  = c & 3;
            uint4 va = *(const uint4*)(X + (size_t)(rBase + row) * E_DIM + kOff + cc * 8);
            unsigned int* da = AsW + row * 18 + cc * 4;
            da[0] = va.x; da[1] = va.y; da[2] = va.z; da[3] = va.w;
            uint4 vb = *(const uint4*)(W + (size_t)(oBase + row) * E_DIM + kOff + cc * 8);
            unsigned int* db = BsW + row * 18 + cc * 4;
            db[0] = vb.x; db[1] = vb.y; db[2] = vb.z; db[3] = vb.w;
        }
        if (kb + 1 < E_DIM / 32) {  // hint next A panel into cache (global_prefetch_b8)
            __builtin_prefetch(X + (size_t)(rBase + (tid >> 1)) * E_DIM + kOff + 32, 0, 1);
        }
        __syncthreads();

        AFrag af[4];
#pragma unroll
        for (int mt = 0; mt < 4; ++mt)
            load_frag(af[mt], AsU, 18, wm * 64 + mt * 16, 0, lane);
        AFrag bf[2];
#pragma unroll
        for (int nt = 0; nt < 2; ++nt)
            load_frag(bf[nt], BsU, 18, wn * 32 + nt * 16, 0, lane);
#pragma unroll
        for (int mt = 0; mt < 4; ++mt)
#pragma unroll
            for (int nt = 0; nt < 2; ++nt)
                acc[mt][nt].v = wmma_bf16(af[mt], bf[nt], acc[mt][nt].v);
        __syncthreads();
    }

    // epilogue: bias + scale, scatter per mode
#pragma unroll
    for (int mt = 0; mt < 4; ++mt) {
#pragma unroll
        for (int nt = 0; nt < 2; ++nt) {
#pragma unroll
            for (int r = 0; r < 8; ++r) {
                const int m = r + ((lane >> 4) << 3);
                const int n = lane & 15;
                const int rrow = rBase + wm * 64 + mt * 16 + m;
                const int o    = oBase + wn * 32 + nt * 16 + n;
                float val = (acc[mt][nt].f[r] + bias[o]) * scale;
                if (MODE == 3) {
                    ((float*)out)[(size_t)rrow * E_DIM + o] = val;
                } else {
                    const int t = rrow >> 2, b = rrow & 3;
                    const int h = o >> 6,    d = o & 63;
                    const int head = b * H_NUM + h;
                    size_t idx;
                    if (MODE == 2)  // V transposed: [head][d][t]
                        idx = ((size_t)head * D_HEAD + d) * T_DIM + t;
                    else            // Q / K: [head][t][d]
                        idx = ((size_t)head * T_DIM + t) * D_HEAD + d;
                    ((unsigned short*)out)[idx] = f2bf(val);
                }
            }
        }
    }
}

// ---------------- kernel 3: flash attention (per head, 128 q-rows / block) ----
__global__ __launch_bounds__(256) void attn_kernel(const unsigned short* __restrict__ Q,
                                                   const unsigned short* __restrict__ K,
                                                   const unsigned short* __restrict__ Vt,
                                                   unsigned short* __restrict__ Xattn) {
    __shared__ unsigned short Qs [128 * 72];   // q rows x 64 d (+8 pad)
    __shared__ unsigned short Ks [ 64 * 72];   // key rows x 64 d
    __shared__ unsigned short Vts[ 64 * 72];   // d rows x 64 keys
    __shared__ float          Sf [128 * 68];   // fp32 scores
    __shared__ unsigned short Ps [128 * 72];   // bf16 probabilities
    __shared__ float          corrS[128];
    __shared__ float          linvS[128];

    const int tid  = threadIdx.x;
    const int lane = tid & 31;
    const int w    = tid >> 5;
    const int wm   = w >> 1;                  // 0..3 : 32-row slice
    const int wn   = w & 1;                   // 0..1 : 32-col slice
    const int head  = blockIdx.y;             // = b*16 + h
    const int qBase = blockIdx.x * 128;
    const size_t hOff = (size_t)head * T_DIM * D_HEAD;

    unsigned int* QsW  = (unsigned int*)Qs;
    unsigned int* KsW  = (unsigned int*)Ks;
    unsigned int* VtsW = (unsigned int*)Vts;
    const unsigned int* QsU  = (const unsigned int*)Qs;
    const unsigned int* KsU  = (const unsigned int*)Ks;
    const unsigned int* VtsU = (const unsigned int*)Vts;
    const unsigned int* PsU  = (const unsigned int*)Ps;

    // stage Q tile: 128 rows x 64 bf16 = 1024 x 16B chunks
#pragma unroll
    for (int it = 0; it < 4; ++it) {
        const int c = tid + it * 256;
        const int row = c >> 3, cc = c & 7;
        uint4 v = *(const uint4*)(Q + hOff + (size_t)(qBase + row) * D_HEAD + cc * 8);
        unsigned int* d = QsW + row * 36 + cc * 4;
        d[0] = v.x; d[1] = v.y; d[2] = v.z; d[3] = v.w;
    }

    float mrow = -1e30f, lrow = 0.0f;         // per-row stats, tid<128 owns row tid
    CFrag acc[2][2];
#pragma unroll
    for (int mt = 0; mt < 2; ++mt)
#pragma unroll
        for (int nt = 0; nt < 2; ++nt)
#pragma unroll
            for (int r = 0; r < 8; ++r) acc[mt][nt].f[r] = 0.0f;

    for (int kb = 0; kb < T_DIM / 64; ++kb) {
        const int kk0 = kb * 64;
        // stage K block (64 keys x 64 d) and Vt block (64 d x 64 keys)
#pragma unroll
        for (int it = 0; it < 2; ++it) {
            const int c = tid + it * 256;
            const int row = c >> 3, cc = c & 7;
            uint4 vk = *(const uint4*)(K + hOff + (size_t)(kk0 + row) * D_HEAD + cc * 8);
            unsigned int* dk = KsW + row * 36 + cc * 4;
            dk[0] = vk.x; dk[1] = vk.y; dk[2] = vk.z; dk[3] = vk.w;
            uint4 vv = *(const uint4*)(Vt + hOff + (size_t)row * T_DIM + kk0 + cc * 8);
            unsigned int* dv = VtsW + row * 36 + cc * 4;
            dv[0] = vv.x; dv[1] = vv.y; dv[2] = vv.z; dv[3] = vv.w;
        }
        __syncthreads();

        // S = Q . K^T  (wave owns 32x32 of the 128x64 score tile)
        CFrag sfr[2][2];
#pragma unroll
        for (int mt = 0; mt < 2; ++mt)
#pragma unroll
            for (int nt = 0; nt < 2; ++nt)
#pragma unroll
                for (int r = 0; r < 8; ++r) sfr[mt][nt].f[r] = 0.0f;
#pragma unroll
        for (int kk = 0; kk < 2; ++kk) {
            AFrag aq[2], bk[2];
#pragma unroll
            for (int mt = 0; mt < 2; ++mt)
                load_frag(aq[mt], QsU, 36, wm * 32 + mt * 16, kk * 16, lane);
#pragma unroll
            for (int nt = 0; nt < 2; ++nt)
                load_frag(bk[nt], KsU, 36, wn * 32 + nt * 16, kk * 16, lane);
#pragma unroll
            for (int mt = 0; mt < 2; ++mt)
#pragma unroll
                for (int nt = 0; nt < 2; ++nt)
                    sfr[mt][nt].v = wmma_bf16(aq[mt], bk[nt], sfr[mt][nt].v);
        }
#pragma unroll
        for (int mt = 0; mt < 2; ++mt)
#pragma unroll
            for (int nt = 0; nt < 2; ++nt)
#pragma unroll
                for (int r = 0; r < 8; ++r) {
                    const int m = wm * 32 + mt * 16 + r + ((lane >> 4) << 3);
                    const int n = wn * 32 + nt * 16 + (lane & 15);
                    Sf[m * 68 + n] = sfr[mt][nt].f[r];
                }
        __syncthreads();

        // online softmax, one thread per q-row
        if (tid < 128) {
            const float* srow = Sf + tid * 68;
            float bm = -1e30f;
#pragma unroll 8
            for (int j = 0; j < 64; ++j) bm = fmaxf(bm, srow[j]);
            const float nm   = fmaxf(mrow, bm);
            const float corr = __expf(mrow - nm);
            float s = 0.0f;
            unsigned short* prow = Ps + tid * 72;
#pragma unroll 8
            for (int j = 0; j < 64; ++j) {
                float p = __expf(srow[j] - nm);
                prow[j] = f2bf(p);
                s += p;
            }
            lrow = lrow * corr + s;
            mrow = nm;
            corrS[tid] = corr;
        }
        __syncthreads();

        // rescale running output, then O += P . V
#pragma unroll
        for (int mt = 0; mt < 2; ++mt)
#pragma unroll
            for (int nt = 0; nt < 2; ++nt)
#pragma unroll
                for (int r = 0; r < 8; ++r) {
                    const int m = wm * 32 + mt * 16 + r + ((lane >> 4) << 3);
                    acc[mt][nt].f[r] *= corrS[m];
                }
#pragma unroll
        for (int kk = 0; kk < 2; ++kk) {
            AFrag ap[2], bv[2];
#pragma unroll
            for (int mt = 0; mt < 2; ++mt)
                load_frag(ap[mt], PsU, 36, wm * 32 + mt * 16, kk * 16, lane);
#pragma unroll
            for (int nt = 0; nt < 2; ++nt)
                load_frag(bv[nt], VtsU, 36, wn * 32 + nt * 16, kk * 16, lane);
#pragma unroll
            for (int mt = 0; mt < 2; ++mt)
#pragma unroll
                for (int nt = 0; nt < 2; ++nt)
                    acc[mt][nt].v = wmma_bf16(ap[mt], bv[nt], acc[mt][nt].v);
        }
        __syncthreads();
    }

    if (tid < 128) linvS[tid] = 1.0f / lrow;
    __syncthreads();

    const int b = head >> 4, h = head & 15;
#pragma unroll
    for (int mt = 0; mt < 2; ++mt)
#pragma unroll
        for (int nt = 0; nt < 2; ++nt)
#pragma unroll
            for (int r = 0; r < 8; ++r) {
                const int m = wm * 32 + mt * 16 + r + ((lane >> 4) << 3);
                const int n = wn * 32 + nt * 16 + (lane & 15);
                const int t = qBase + m;
                float v = acc[mt][nt].f[r] * linvS[m];
                Xattn[(size_t)(t * B_DIM + b) * E_DIM + h * D_HEAD + n] = f2bf(v);
            }
}

// ---------------- launch ----------------
extern "C" void kernel_launch(void* const* d_in, const int* in_sizes, int n_in,
                              void* d_out, int out_size, void* d_ws, size_t ws_size,
                              hipStream_t stream) {
    const float* query = (const float*)d_in[0];
    const float* Wq = (const float*)d_in[1];
    const float* bq = (const float*)d_in[2];
    const float* Wk = (const float*)d_in[3];
    const float* bk = (const float*)d_in[4];
    const float* Wv = (const float*)d_in[5];
    const float* bv = (const float*)d_in[6];
    const float* Wo = (const float*)d_in[7];
    const float* bo = (const float*)d_in[8];

    char* ws = (char*)d_ws;
    const size_t WSZ = (size_t)E_DIM * E_DIM * 2;       // 2 MB per weight (bf16)
    const size_t XSZ = (size_t)R_ROWS * E_DIM * 2;      // 16 MB
    unsigned short* WqB = (unsigned short*)(ws + 0 * WSZ);
    unsigned short* WkB = (unsigned short*)(ws + 1 * WSZ);
    unsigned short* WvB = (unsigned short*)(ws + 2 * WSZ);
    unsigned short* WoB = (unsigned short*)(ws + 3 * WSZ);
    unsigned short* Xb  = (unsigned short*)(ws + 4 * WSZ);
    unsigned short* Qb  = (unsigned short*)(ws + 4 * WSZ + 1 * XSZ);
    unsigned short* Kb  = (unsigned short*)(ws + 4 * WSZ + 2 * XSZ);
    unsigned short* Vtb = (unsigned short*)(ws + 4 * WSZ + 3 * XSZ);
    unsigned short* Xa  = (unsigned short*)(ws + 4 * WSZ + 4 * XSZ);

    const int nW = E_DIM * E_DIM;          // 1 M
    const int nX = R_ROWS * E_DIM;         // 8 M
    cvt_bf16<<<(nW + 255) / 256, 256, 0, stream>>>(Wq, WqB, nW);
    cvt_bf16<<<(nW + 255) / 256, 256, 0, stream>>>(Wk, WkB, nW);
    cvt_bf16<<<(nW + 255) / 256, 256, 0, stream>>>(Wv, WvB, nW);
    cvt_bf16<<<(nW + 255) / 256, 256, 0, stream>>>(Wo, WoB, nW);
    cvt_bf16<<<(nX + 255) / 256, 256, 0, stream>>>(query, Xb, nX);

    dim3 gGemm(E_DIM / 128, R_ROWS / 128);   // (8, 64)
    proj_gemm<0><<<gGemm, 256, 0, stream>>>(Xb, WqB, bq, (void*)Qb, SCALING);
    proj_gemm<1><<<gGemm, 256, 0, stream>>>(Xb, WkB, bk, (void*)Kb, 1.0f);
    proj_gemm<2><<<gGemm, 256, 0, stream>>>(Xb, WvB, bv, (void*)Vtb, 1.0f);

    dim3 gAttn(T_DIM / 128, B_DIM * H_NUM);  // (16, 64)
    attn_kernel<<<gAttn, 256, 0, stream>>>(Qb, Kb, Vtb, Xa);

    proj_gemm<3><<<gGemm, 256, 0, stream>>>(Xa, WoB, bo, d_out, 1.0f);
}
